// SnnYoloV3WM_40157944217702
// MI455X (gfx1250) — matrix operand order repaired and checked by reading, not compiled, not run on hardware
//
// SNN-YOLO forward for MI455X (gfx1250). Convolutions are implicit-GEMM on
// V_WMMA_F32_16X16X32_F16 with N-tile register blocking (up to 4 accumulators
// per wave, amortizing the im2col A-gather over 4 WMMAs). LIF spikes are
// stored as f16 ({0,1} is exact in f16) halving the dominant read stream.
// All orchestration is stream-ordered and graph-capture safe.
//
// WMMA register layouts used (CDNA5 ISA 7.12.2):
//  A (16x32 f16): lane l row M=l&15; halfs j=0..7 -> K = (l<16?0:8)+j,
//                 halfs j=8..15 -> K = 16 + (l<16?0:8) + (j-8).
//  B (32x16 f16): lane l col N=l&15; halfs j=0..15 -> same K as A (pack_w_kernel mirrors the A gather)
//  C/D (16x16 f32): VGPR r, lane l -> M = r + (l<16?0:8), N = l&15.
#include <hip/hip_runtime.h>

typedef __attribute__((ext_vector_type(16))) _Float16 v16h;
typedef __attribute__((ext_vector_type(8)))  float    v8f;

static __device__ __forceinline__ float heav(float v) { return v > 0.f ? 1.f : 0.f; }

// ---------------- LIF scan (snntorch Leaky, reset-by-subtraction) -----------
// Emits f16 spikes: values are exactly {0,1}, so f16 is lossless and halves
// the bandwidth of the largest tensors in the graph.
__global__ void lif_kernel(const float* __restrict__ x, _Float16* __restrict__ spk,
                           const float* __restrict__ beta_p, const float* __restrict__ th_p,
                           long N, int T) {
  long i = (long)blockIdx.x * blockDim.x + threadIdx.x;
  if (i >= N) return;
  float b = beta_p[0];
  b = b < 0.f ? 0.f : (b > 1.f ? 1.f : b);
  float th = th_p[0];
  float mem = 0.f;
  for (int t = 0; t < T; ++t) {
    float reset = heav(mem - th);
    mem = b * mem + x[(long)t * N + i] - reset * th;
    spk[(long)t * N + i] = (_Float16)heav(mem - th);
  }
}

// ---------------- weight pre-pack into per-lane B fragments -----------------
__global__ void pack_w_kernel(const float* __restrict__ w, _Float16* __restrict__ pack,
                              int Cout, int Ktot, int Kch, int Ntiles) {
  int idx = blockIdx.x * blockDim.x + threadIdx.x;
  int total = Ntiles * Kch * 512;
  if (idx >= total) return;
  int j    = idx & 15;
  int lane = (idx >> 4) & 31;
  int kc   = (idx >> 9) % Kch;
  int nt   = idx / (Kch * 512);
  int cout = nt * 16 + (lane & 15);
  int k    = kc * 32 + ((lane < 16) ? 0 : 8) + (j < 8 ? j : j + 8);
  float v = 0.f;
  if (cout < Cout && k < Ktot) v = w[(long)cout * Ktot + k];
  pack[idx] = (_Float16)v;
}

// ---------------- implicit-GEMM conv --------------------------------------
// One wave per (image, 16-pixel M-tile, NTB-wide group of 16-cout N-tiles).
// The A fragment (im2col gather) is built once per K-chunk and reused by NTB
// WMMAs against NTB pre-packed B fragments.
template <int KK, int NTB, typename TIN>
__global__ void conv_wmma_kernel(const TIN* __restrict__ inA, long sA, int CA,
                                 const TIN* __restrict__ inB, long sB, int CB,
                                 const _Float16* __restrict__ wpack,
                                 const float* __restrict__ bias,
                                 float* __restrict__ out, long sO,
                                 int NI, int H, int W, int Ho, int Wo,
                                 int Cout, int stride, int pad) {
  const int K2   = KK * KK;
  const int Ktot = (CA + CB) * K2;
  const int Kch  = (Ktot + 31) >> 5;
  const int Mtot = Ho * Wo;
  const int Mt   = (Mtot + 15) >> 4;
  const int Nt   = (Cout + 15) >> 4;
  const int Ngrp = Nt / NTB;             // host guarantees divisibility
  long wid = ((long)blockIdx.x * blockDim.x + threadIdx.x) >> 5;
  int lane = threadIdx.x & 31;
  long total = (long)NI * Mt * Ngrp;
  if (wid >= total) return;              // wave-uniform: EXEC stays all-ones
  int  grp = (int)(wid % Ngrp);
  long r0  = wid / Ngrp;
  int  mt  = (int)(r0 % Mt);
  int  n   = (int)(r0 / Mt);
  int  nt0 = grp * NTB;

  int m  = mt * 16 + (lane & 15);
  int mc = m < Mtot ? m : (Mtot - 1);    // clamp OOB rows (store-masked later)
  int oy = mc / Wo;
  int ox = mc - oy * Wo;
  int iy0 = oy * stride - pad;
  int ix0 = ox * stride - pad;
  int koff = (lane < 16) ? 0 : 8;

  const TIN* baseA = inA + (long)n * sA;
  const TIN* baseB = inB + (long)n * sB;

  v8f acc[NTB];
#pragma unroll
  for (int i = 0; i < NTB; ++i) acc[i] = (v8f){};

  const _Float16* wbase = wpack + (long)nt0 * Kch * 512 + lane * 16;
  for (int kc = 0; kc < Kch; ++kc) {
    __builtin_prefetch(wbase + (long)(kc + 1) * 512, 0, 0);  // next weight chunk
    v16h a;
#pragma unroll
    for (int j = 0; j < 16; ++j) {
      int k = kc * 32 + koff + (j < 8 ? j : j + 8);
      float v = 0.f;
      if (k < Ktot) {
        int c  = k / K2;
        int rr = k - c * K2;
        int kh = rr / KK;
        int kw = rr - kh * KK;
        int iy = iy0 + kh;
        int ix = ix0 + kw;
        if (iy >= 0 && iy < H && ix >= 0 && ix < W) {
          const TIN* p = (c < CA)
              ? (baseA + ((long)c * H + iy) * W + ix)
              : (baseB + ((long)(c - CA) * H + iy) * W + ix);
          v = (float)(*p);
        }
      }
      a[j] = (_Float16)v;
    }
#pragma unroll
    for (int i = 0; i < NTB; ++i) {
      v16h bfr = *(const v16h*)(wbase + ((long)i * Kch + kc) * 512);
      acc[i] = __builtin_amdgcn_wmma_f32_16x16x32_f16(false, a, false, bfr, (short)0,
                                                      acc[i], false, false);
    }
  }
  int mbase = mt * 16 + ((lane < 16) ? 0 : 8);
#pragma unroll
  for (int i = 0; i < NTB; ++i) {
    int cout = (nt0 + i) * 16 + (lane & 15);
    if (cout < Cout) {
      float bv = bias ? bias[cout] : 0.f;
      float* baseO = out + (long)n * sO + (long)cout * Mtot;
#pragma unroll
      for (int r = 0; r < 8; ++r) {
        int mm = mbase + r;
        if (mm < Mtot) baseO[mm] = acc[i][r] + bv;
      }
    }
  }
}

// ---------------- BatchNorm (training-mode batch stats, gamma only) ---------
__global__ void bn_stats_kernel(const float* __restrict__ x, float* __restrict__ stats,
                                int C, long HW, int Bn, long imgStride, long tStride) {
  int tc = blockIdx.x;
  int t = tc / C, c = tc % C;
  const float* base = x + (long)t * tStride + (long)c * HW;
  float s1 = 0.f, s2 = 0.f;
  long cnt = (long)Bn * HW;
  for (long i = threadIdx.x; i < cnt; i += blockDim.x) {
    long b = i / HW, off = i - b * HW;
    float v = base[b * imgStride + off];
    s1 += v; s2 += v * v;
  }
  __shared__ float sh1[256];
  __shared__ float sh2[256];
  sh1[threadIdx.x] = s1; sh2[threadIdx.x] = s2;
  __syncthreads();
  for (int s = 128; s > 0; s >>= 1) {
    if ((int)threadIdx.x < s) {
      sh1[threadIdx.x] += sh1[threadIdx.x + s];
      sh2[threadIdx.x] += sh2[threadIdx.x + s];
    }
    __syncthreads();
  }
  if (threadIdx.x == 0) {
    float fc = (float)cnt;
    float mu = sh1[0] / fc;
    stats[2 * tc]     = mu;
    stats[2 * tc + 1] = sh2[0] / fc - mu * mu;
  }
}

__global__ void bn_apply_kernel(float* __restrict__ x, const float* __restrict__ stats,
                                const float* __restrict__ gamma, int T, int Bn, int C, long HW) {
  long idx = (long)blockIdx.x * blockDim.x + threadIdx.x;
  long total = (long)T * Bn * C * HW;
  if (idx >= total) return;
  long r = idx / HW;
  int c = (int)(r % C); r /= C;
  r /= Bn;                       // skip b
  int t = (int)r;
  int tc = t * C + c;
  float mu = stats[2 * tc], var = stats[2 * tc + 1];
  x[idx] = (x[idx] - mu) * rsqrtf(var + 1e-5f) * gamma[tc];
}

// ---------------- maxpool 3x3 s2 p1 ----------------------------------------
__global__ void maxpool_kernel(const float* __restrict__ in, float* __restrict__ out,
                               int NI, int C, int H, int W, int Ho, int Wo) {
  long idx = (long)blockIdx.x * blockDim.x + threadIdx.x;
  long total = (long)NI * C * Ho * Wo;
  if (idx >= total) return;
  int ox = (int)(idx % Wo); long r = idx / Wo;
  int oy = (int)(r % Ho); r /= Ho;
  int c = (int)(r % C); int n = (int)(r / C);
  const float* base = in + ((long)n * C + c) * H * W;
  float m = -__builtin_inff();
  int ys = oy * 2 - 1, xs = ox * 2 - 1;
  for (int dy = 0; dy < 3; ++dy) {
    int y = ys + dy; if (y < 0 || y >= H) continue;
    for (int dx = 0; dx < 3; ++dx) {
      int x = xs + dx; if (x < 0 || x >= W) continue;
      float v = base[(long)y * W + x];
      m = v > m ? v : m;
    }
  }
  out[idx] = m;
}

// ---------------- elementwise helpers --------------------------------------
__global__ void add_kernel(const float* __restrict__ a, const float* __restrict__ b,
                           float* __restrict__ o, long N) {
  long i = (long)blockIdx.x * blockDim.x + threadIdx.x;
  if (i < N) o[i] = a[i] + b[i];
}

__global__ void copych_kernel(const float* __restrict__ src, float* __restrict__ dst,
                              int NI, int Csrc, long HW, int Cdst, int coff) {
  long i = (long)blockIdx.x * blockDim.x + threadIdx.x;
  long total = (long)NI * Csrc * HW;
  if (i >= total) return;
  long p = i % HW; long r = i / HW;
  int c = (int)(r % Csrc); int n = (int)(r / Csrc);
  dst[(((long)n * Cdst) + coff + c) * HW + p] = src[i];
}

// out[b][t] = y[t][b] + r[t][b]
__global__ void final_out_kernel(const float* __restrict__ y, const float* __restrict__ r,
                                 float* __restrict__ o, int T, int Bn, long chw) {
  long idx = (long)blockIdx.x * blockDim.x + threadIdx.x;
  long total = (long)T * Bn * chw;
  if (idx >= total) return;
  long i = idx % chw;
  long rr = idx / chw;
  int b = (int)(rr % Bn);
  int t = (int)(rr / Bn);
  o[((long)b * T + t) * chw + i] = y[idx] + r[idx];
}

// ================= host orchestration =======================================
static inline int blocks_for(long n) { return (int)((n + 255) / 256); }

template <typename TIN>
static void conv_run(hipStream_t st, _Float16* pack,
                     const TIN* inA, long sA, int CA,
                     const TIN* inB, long sB, int CB,
                     const float* w, const float* bias,
                     float* out, long sO,
                     int NI, int H, int W, int KK, int stride, int pad, int Cout) {
  int Ho = (H + 2 * pad - KK) / stride + 1;
  int Wo = (W + 2 * pad - KK) / stride + 1;
  int Ktot = (CA + CB) * KK * KK;
  int Kch = (Ktot + 31) / 32;
  int Nt = (Cout + 15) / 16;
  long ptotal = (long)Nt * Kch * 512;
  pack_w_kernel<<<blocks_for(ptotal), 256, 0, st>>>(w, pack, Cout, Ktot, Kch, Nt);
  long Mtot = (long)Ho * Wo;
  long Mt = (Mtot + 15) / 16;
  int NTB = (Nt % 4 == 0) ? 4 : (Nt % 2 == 0) ? 2 : 1;
  long waves = (long)NI * Mt * (Nt / NTB);
  int blocks = (int)((waves * 32 + 255) / 256);
#define LAUNCH(KKv, NTBv)                                                            \
  conv_wmma_kernel<KKv, NTBv, TIN><<<blocks, 256, 0, st>>>(                          \
      inA, sA, CA, inB, sB, CB, pack, bias, out, sO, NI, H, W, Ho, Wo, Cout, stride, pad)
  if (KK == 3) {
    if (NTB == 4) LAUNCH(3, 4);
    else if (NTB == 2) LAUNCH(3, 2);
    else LAUNCH(3, 1);
  } else {
    if (NTB == 4) LAUNCH(1, 4);
    else if (NTB == 2) LAUNCH(1, 2);
    else LAUNCH(1, 1);
  }
#undef LAUNCH
}

static void bn_run(hipStream_t st, float* x, float* stats, const float* gamma,
                   int T, int Bn, int C, long HW) {
  bn_stats_kernel<<<T * C, 256, 0, st>>>(x, stats, C, HW, Bn, (long)C * HW, (long)Bn * C * HW);
  long total = (long)T * Bn * C * HW;
  bn_apply_kernel<<<blocks_for(total), 256, 0, st>>>(x, stats, gamma, T, Bn, C, HW);
}

struct LcbP { const float *beta, *th, *w, *b, *g; };

static void run_lcb(hipStream_t st, const float* in, _Float16* spkbuf, float* outbuf,
                    _Float16* pack, float* stats, const LcbP& p,
                    int cin, int cout, int H, int stride) {
  const int T = 4, Bn = 4, TB = 16;
  long Nin = (long)cin * H * H;
  lif_kernel<<<blocks_for(Bn * Nin), 256, 0, st>>>(in, spkbuf, p.beta, p.th, Bn * Nin, T);
  int Ho = (H + 2 - 3) / stride + 1;
  conv_run<_Float16>(st, pack, spkbuf, Nin, cin, (const _Float16*)nullptr, 0, 0, p.w, p.b,
                     outbuf, (long)cout * Ho * Ho, TB, H, H, 3, stride, 1, cout);
  bn_run(st, outbuf, stats, p.g, T, Bn, cout, (long)Ho * Ho);
}

static void run_m2(hipStream_t st, float* A, float* B, float* C, float* D, float* E,
                   float* Fb, float* G, float* Hs, _Float16* pack, float* stats,
                   const LcbP& L0, const LcbP& L1, const LcbP& R,
                   const LcbP& M1, const LcbP& M2, int cin, int cout, int H) {
  const int TB = 16;
  int Ho = H / 2;
  long HWo = (long)Ho * Ho;
  run_lcb(st, A, (_Float16*)B, C, pack, stats, L0, cin, cout, H, 2);   // xl (stride 2)
  run_lcb(st, C, (_Float16*)D, E, pack, stats, L1, cout, cout, Ho, 1); // xl = lcb(xl)
  maxpool_kernel<<<blocks_for((long)TB * cin * HWo), 256, 0, st>>>(A, G, TB, cin, H, H, Ho, Ho);
  run_lcb(st, G, (_Float16*)Hs, D, pack, stats, R, cin, cout - cin, Ho, 1); // xr0
  copych_kernel<<<blocks_for((long)TB * cin * HWo), 256, 0, st>>>(G, Fb, TB, cin, HWo, cout, 0);
  copych_kernel<<<blocks_for((long)TB * (cout - cin) * HWo), 256, 0, st>>>(D, Fb, TB, cout - cin,
                                                                           HWo, cout, cin);
  long Nz = (long)TB * cout * HWo;
  add_kernel<<<blocks_for(Nz), 256, 0, st>>>(E, Fb, C, Nz);            // z = xl + cat(xr,xr0)
  run_lcb(st, C, (_Float16*)D, G, pack, stats, M1, cout, cout / 2, Ho, 1); // ms.lcb1
  run_lcb(st, G, (_Float16*)Hs, E, pack, stats, M2, cout / 2, cout, Ho, 1); // ms.lcb2
  add_kernel<<<blocks_for(Nz), 256, 0, st>>>(E, C, A, Nz);             // ms residual -> A
}

static void run_tcnn(hipStream_t st, const float* x, long x_tStride, long x_imgStride,
                     float* sOut, float* hTmp, _Float16* pack, float* stats,
                     const float* c1w, const float* c1b, const float* cfw, const float* cfb,
                     const float* g, int cin, int cout, int H, int KK, int pad) {
  const int T = 4, Bn = 4;
  long HW = (long)H * H;
  long NimgO = (long)cout * HW;
  long tO = Bn * NimgO;
  long NimgH = (long)cin * HW;
  conv_run<float>(st, pack, x, x_imgStride, cin, (const float*)nullptr, 0, 0, c1w, c1b,
                  sOut, NimgO, Bn, H, H, KK, 1, pad, cout);            // s0 = conv1(x_0)
  for (int t = 1; t < T; ++t) {
    conv_run<float>(st, pack, x + (long)t * x_tStride, x_imgStride, cin,  // convf(cat(x_t, s_{t-1}))
                    sOut + (long)(t - 1) * tO, NimgO, cout,
                    cfw, cfb, hTmp, NimgH, Bn, H, H, KK, 1, pad, cin);
    conv_run<float>(st, pack, hTmp, NimgH, cin, (const float*)nullptr, 0, 0, c1w, c1b,
                    sOut + (long)t * tO, NimgO, Bn, H, H, KK, 1, pad, cout); // s_t = conv1(h)
  }
  bn_run(st, sOut, stats, g, T, Bn, cout, HW);
}

extern "C" void kernel_launch(void* const* d_in, const int* in_sizes, int n_in,
                              void* d_out, int out_size, void* d_ws, size_t ws_size,
                              hipStream_t stream) {
  (void)in_sizes; (void)n_in; (void)out_size; (void)ws_size;
  const float* X = (const float*)d_in[0];
#define FP(i) ((const float*)d_in[i])
  const float *l0_c1w = FP(1), *l0_cfw = FP(2), *l0_g = FP(3), *l0_c1b = FP(4), *l0_cfb = FP(5);
  auto lcb_at = [&](int i) { return LcbP{FP(i), FP(i + 1), FP(i + 2), FP(i + 3), FP(i + 4)}; };
  LcbP m1L0 = lcb_at(6),  m1L1 = lcb_at(11), m1R = lcb_at(16), m1A = lcb_at(21), m1B = lcb_at(26);
  LcbP m2L0 = lcb_at(31), m2L1 = lcb_at(36), m2R = lcb_at(41), m2A = lcb_at(46), m2B = lcb_at(51);
  const float *ra_c1w = FP(56), *ra_cfw = FP(57), *ra_g = FP(58);
  const float *rb_c1w = FP(59), *rb_cfw = FP(60), *rb_g = FP(61);
#undef FP

  // workspace arena (float offsets); total ~306.3 MB
  float* ws = (float*)d_ws;
  float* A  = ws + 0L;          // 16.78M  y0 / m2 in-out
  float* B  = ws + 16777216L;   // 16.78M  big spike buffer (used as f16)
  float* C  = ws + 33554432L;   // 8.39M
  float* D  = ws + 41943040L;   // 8.39M
  float* E  = ws + 50331648L;   // 8.39M
  float* Fb = ws + 58720256L;   // 8.39M
  float* G  = ws + 67108864L;   // 4.19M
  float* Hs = ws + 71303168L;   // 4.19M
  float* HT = ws + 75497472L;   // 1.05M  tcnn hidden temp
  float* ST = ws + 76546048L;   // 1K     BN stats
  _Float16* PK = (_Float16*)(ws + 76547072L);  // 64K halfs, 32B aligned

  // layer0: TCNN 3->16 k3 p1 on external x [B,T,3,256,256]
  run_tcnn(stream, X, 196608L, 786432L, A, HT, PK, ST,
           l0_c1w, l0_c1b, l0_cfw, l0_cfb, l0_g, 3, 16, 256, 3, 1);
  // layer1: m2 16->32 @256 -> [T,B,32,128,128] in A
  run_m2(stream, A, B, C, D, E, Fb, G, Hs, PK, ST, m1L0, m1L1, m1R, m1A, m1B, 16, 32, 256);
  // layer2: m2 32->64 @128 -> [T,B,64,64,64] in A
  run_m2(stream, A, B, C, D, E, Fb, G, Hs, PK, ST, m2L0, m2L1, m2R, m2A, m2B, 32, 64, 128);
  // res1_a: TCNN 64->32 k1 p0 (no bias) on A -> C
  run_tcnn(stream, A, 4L * 64 * 4096, 64L * 4096, C, D, PK, ST,
           ra_c1w, nullptr, ra_cfw, nullptr, ra_g, 64, 32, 64, 1, 0);
  // res1_b: TCNN 32->64 k3 p1 (no bias) on C -> E
  run_tcnn(stream, C, 4L * 32 * 4096, 32L * 4096, E, D, PK, ST,
           rb_c1w, nullptr, rb_cfw, nullptr, rb_g, 32, 64, 64, 3, 1);
  // out = y + r, [T,B,...] -> [B,T,...]
  long total = 4L * 4 * 64 * 4096;
  final_out_kernel<<<blocks_for(total), 256, 0, stream>>>(A, E, (float*)d_out, 4, 4,
                                                          64L * 4096);
}
